// Attention_42365557407945
// MI455X (gfx1250) — matrix operand order repaired and checked by reading, not compiled, not run on hardware
//
#include <hip/hip_runtime.h>

// ---------------------------------------------------------------------------
// Multi-head attention (B=8, N=1024, C=768, H=12, HD=64) for x and y,
// shared weights. GEMMs on V_WMMA_F32_16X16X32_F16; attention K/V tiles
// staged by the Tensor Data Mover (tensor_load_to_lds + s_wait_tensorcnt)
// with double buffering.
// ---------------------------------------------------------------------------

typedef _Float16 h16;
typedef _Float16 v16h __attribute__((ext_vector_type(16)));
typedef _Float16 v8h  __attribute__((ext_vector_type(8)));
typedef _Float16 v4h  __attribute__((ext_vector_type(4)));
typedef float    v8f  __attribute__((ext_vector_type(8)));
typedef unsigned int v4u __attribute__((ext_vector_type(4)));
typedef int      v8i  __attribute__((ext_vector_type(8)));
typedef int      v4i  __attribute__((ext_vector_type(4)));

#define WMMA_F16(a, b, c) \
  __builtin_amdgcn_wmma_f32_16x16x32_f16(false, (a), false, (b), (short)0, (c), false, false)

// ---- fragment loaders (wave32 layouts per CDNA5 ISA 7.12.2) ----------------
static __device__ __forceinline__ v16h load_frag_a(const h16* tile, int row0,
                                                   int k0, int ldk, int lane) {
  const int hl   = lane >> 4;
  const int mrow = lane & 15;
  const h16* p = tile + (row0 + mrow) * ldk;
  v8h lo = *(const v8h*)(p + k0 + hl * 8);
  v8h hi = *(const v8h*)(p + k0 + 16 + hl * 8);
  return __builtin_shufflevector(lo, hi, 0, 1, 2, 3, 4, 5, 6, 7,
                                 8, 9, 10, 11, 12, 13, 14, 15);
}

static __device__ __forceinline__ v16h load_frag_b(const h16* tile, int col0,
                                                   int k0, int ldk, int lane) {
  const h16* p = tile + (col0 + (lane & 15)) * ldk + k0 + ((lane >> 4) << 3);
  v8h lo = *(const v8h*)(p);
  v8h hi = *(const v8h*)(p + 16);
  return __builtin_shufflevector(lo, hi, 0, 1, 2, 3, 4, 5, 6, 7,
                                 8, 9, 10, 11, 12, 13, 14, 15);
}

// ---- Tensor Data Mover: 2D tile load global -> LDS -------------------------
// D# layout per CDNA5 ISA ch.8 / CDNA5 ISA.
// 16-bit elements (data_size code 1). LDS padding: pad_amount DWORDs inserted
// after every pad_interval-coded DWORD span (code 4 => 32 DW = 64 halves,
// code 3 pad => 4 DW = 8 halves -> padded row stride 72 halves).
static __device__ __forceinline__ void tdm_load_2d(unsigned lds_off,
                                                   const void* gptr,
                                                   unsigned tensor_d0,
                                                   unsigned tensor_d1,
                                                   unsigned stride0,
                                                   unsigned tile0,
                                                   unsigned tile1,
                                                   unsigned pad_interval,
                                                   unsigned pad_amount) {
  unsigned long long ga = (unsigned long long)gptr;
  v4u g0;
  g0.x = 1u;                                         // count=1, user descriptor
  g0.y = lds_off;                                    // lds_addr (bytes)
  g0.z = (unsigned)(ga & 0xFFFFFFFFu);               // global_addr[31:0]
  g0.w = (unsigned)((ga >> 32) & 0x01FFFFFFu)        // global_addr[56:32]
         | (2u << 30);                               // type = 2 ("image")
  v8i g1;
  g1[0] = (int)((1u << 16)                           // data_size = 2 bytes
                | (1u << 20)                         // pad_enable
                | (pad_interval << 22)
                | (pad_amount << 25));
  g1[1] = (int)((tensor_d0 & 0xFFFFu) << 16);        // tensor_dim0[15:0]
  g1[2] = (int)((tensor_d0 >> 16) |                  // tensor_dim0[31:16]
                ((tensor_d1 & 0xFFFFu) << 16));      // tensor_dim1[15:0]
  g1[3] = (int)((tensor_d1 >> 16) |                  // tensor_dim1[31:16]
                ((tile0 & 0xFFFFu) << 16));          // tile_dim0
  g1[4] = (int)(tile1 & 0xFFFFu);                    // tile_dim1 (tile_dim2=0)
  g1[5] = (int)stride0;                              // tensor_dim0_stride[31:0]
  g1[6] = 0;                                         // stride0 hi / stride1 lo
  g1[7] = 0;
  v4i z4 = {0, 0, 0, 0};
#if __clang_major__ >= 23
  v8i z8 = {0, 0, 0, 0, 0, 0, 0, 0};
  __builtin_amdgcn_tensor_load_to_lds(g0, g1, z4, z4, z8, 0);
#else
  __builtin_amdgcn_tensor_load_to_lds(g0, g1, z4, z4, 0);
#endif
}

#define LDSOFF(p) ((unsigned)(size_t)(p))

// ---- fp32 -> f16 conversion (vectorized x4) --------------------------------
__global__ void cvt4_f32_f16(const float* __restrict__ src,
                             h16* __restrict__ dst, int n4) {
  int i = blockIdx.x * 256 + threadIdx.x;
  if (i < n4) {
    float4 v = ((const float4*)src)[i];
    v4h o = { (h16)v.x, (h16)v.y, (h16)v.z, (h16)v.w };
    ((v4h*)dst)[i] = o;
  }
}

// ---- QKV GEMM: [8192,768]f16 x [2304,768]f16(B^T) -> Q,K,V^T f16 -----------
__global__ __launch_bounds__(128) void qkv_gemm(const h16* __restrict__ X,
                                                const h16* __restrict__ W,
                                                h16* __restrict__ Qb,
                                                h16* __restrict__ Kb,
                                                h16* __restrict__ Vtb) {
  __shared__ __align__(16) h16 As[64 * 40];
  __shared__ __align__(16) h16 Bs[64 * 40];
  const int tid = threadIdx.x, lane = tid & 31, wave = tid >> 5;
  const int wm = (wave >> 1) * 32, wn = (wave & 1) * 32;
  const int mb = blockIdx.x * 64;
  const int nb = blockIdx.y * 64;
  const int lr = tid >> 1, lk = (tid & 1) * 16;

  v8f acc[2][2] = {};
  for (int k0 = 0; k0 < 768; k0 += 32) {
    const v8h* ga = (const v8h*)(X + (size_t)(mb + lr) * 768 + k0 + lk);
    *(v8h*)(As + lr * 40 + lk)     = ga[0];
    *(v8h*)(As + lr * 40 + lk + 8) = ga[1];
    const v8h* gb = (const v8h*)(W + (size_t)(nb + lr) * 768 + k0 + lk);
    *(v8h*)(Bs + lr * 40 + lk)     = gb[0];
    *(v8h*)(Bs + lr * 40 + lk + 8) = gb[1];
    if (k0 + 32 < 768) {
      __builtin_prefetch(X + (size_t)(mb + lr) * 768 + k0 + 32 + lk, 0, 1);
      __builtin_prefetch(W + (size_t)(nb + lr) * 768 + k0 + 32 + lk, 0, 1);
    }
    __syncthreads();
    v16h a0 = load_frag_a(As, wm, 0, 40, lane);
    v16h a1 = load_frag_a(As, wm + 16, 0, 40, lane);
    v16h b0 = load_frag_b(Bs, wn, 0, 40, lane);
    v16h b1 = load_frag_b(Bs, wn + 16, 0, 40, lane);
    acc[0][0] = WMMA_F16(a0, b0, acc[0][0]);
    acc[0][1] = WMMA_F16(a0, b1, acc[0][1]);
    acc[1][0] = WMMA_F16(a1, b0, acc[1][0]);
    acc[1][1] = WMMA_F16(a1, b1, acc[1][1]);
    __syncthreads();
  }

  const float scale = 0.125f;  // 64^-0.5
  const int hl = lane >> 4, nc = lane & 15;
  for (int af = 0; af < 2; ++af)
    for (int tf = 0; tf < 2; ++tf)
      for (int r = 0; r < 8; ++r) {
        int m = mb + wm + af * 16 + r + hl * 8;
        int j = nb + wn + tf * 16 + nc;
        float v = acc[af][tf][r];
        int t = j / 768, rem = j % 768;
        int hh = rem >> 6, d = rem & 63;
        int bi = m >> 10, n = m & 1023;
        size_t bh = (size_t)(bi * 12 + hh);
        if (t == 0)      Qb[bh * 65536 + n * 64 + d] = (h16)(v * scale);
        else if (t == 1) Kb[bh * 65536 + n * 64 + d] = (h16)v;
        else             Vtb[bh * 65536 + d * 1024 + n] = (h16)v;
      }
}

// ---- flash attention per (b,h): TDM-staged, double-buffered K/V ------------
__global__ __launch_bounds__(128) void attn_kernel(const h16* __restrict__ Qb,
                                                   const h16* __restrict__ Kb,
                                                   const h16* __restrict__ Vtb,
                                                   h16* __restrict__ AO) {
  __shared__ __align__(16) h16 Qs[64 * 72];
  __shared__ __align__(16) h16 Ks[2][64 * 72];
  __shared__ __align__(16) h16 Vs[2][64 * 72];   // V^T tiles: [d][n]
  __shared__ __align__(16) h16 Ps[4 * 16 * 72];  // per-wave P tiles
  const int tid = threadIdx.x, lane = tid & 31, w = tid >> 5;
  const int bh = blockIdx.y, rb = blockIdx.x * 64;
  const h16* Qg = Qb + (size_t)bh * 65536;
  const h16* Kg = Kb + (size_t)bh * 65536;
  const h16* Vg = Vtb + (size_t)bh * 65536;

  // Pre-issue: Q block + first K/V tiles (TDM, per-wave scalar op from wave 0).
  if (w == 0) {
    tdm_load_2d(LDSOFF(Qs),    Qg + (size_t)rb * 64, 64, 64, 64,   64, 64, 4, 3);
    tdm_load_2d(LDSOFF(Ks[0]), Kg,                   64, 64, 64,   64, 64, 4, 3);
    tdm_load_2d(LDSOFF(Vs[0]), Vg,                   64, 64, 1024, 64, 64, 4, 3);
  }

  float mrow[8], lrow[8];
  v8f o[4] = {};
  for (int r = 0; r < 8; ++r) { mrow[r] = -3.0e38f; lrow[r] = 0.0f; }

  for (int jb = 0; jb < 16; ++jb) {
    const int cur = jb & 1;
    if (w == 0) {
      if (jb < 15) {
        // issue next tiles into the alternate buffer, then retire current:
        // TDM ops are in-order per wave, so tensorcnt<=2 means the current
        // buffer's loads (and Q on the first pass) have landed in LDS.
        tdm_load_2d(LDSOFF(Ks[cur ^ 1]), Kg + (size_t)(jb + 1) * 64 * 64,
                    64, 64, 64, 64, 64, 4, 3);
        tdm_load_2d(LDSOFF(Vs[cur ^ 1]), Vg + (jb + 1) * 64,
                    64, 64, 1024, 64, 64, 4, 3);
        __builtin_amdgcn_s_wait_tensorcnt(2);
      } else {
        __builtin_amdgcn_s_wait_tensorcnt(0);
      }
    }
    __syncthreads();  // publish TDM-written LDS to all waves
    const h16* Kc = Ks[cur];
    const h16* Vc = Vs[cur];

    // S = Q_blk (16x64 per wave) x K_blk^T (64x64)
    v16h a0 = load_frag_a(Qs, w * 16, 0, 72, lane);
    v16h a1 = load_frag_a(Qs, w * 16, 32, 72, lane);
    v8f sc[4];
    for (int t = 0; t < 4; ++t) {
      v16h b0 = load_frag_b(Kc, t * 16, 0, 72, lane);
      v16h b1 = load_frag_b(Kc, t * 16, 32, 72, lane);
      v8f z = {};
      z = WMMA_F16(a0, b0, z);
      sc[t] = WMMA_F16(a1, b1, z);
    }

    // online softmax (row stats replicated across each 16-lane group)
    for (int r = 0; r < 8; ++r) {
      float mx = fmaxf(fmaxf(sc[0][r], sc[1][r]), fmaxf(sc[2][r], sc[3][r]));
      for (int off = 8; off >= 1; off >>= 1)
        mx = fmaxf(mx, __shfl_xor(mx, off, 16));
      float mnew = fmaxf(mrow[r], mx);
      float corr = __expf(mrow[r] - mnew);
      mrow[r] = mnew;
      float psum = 0.0f;
      for (int t = 0; t < 4; ++t) {
        float p = __expf(sc[t][r] - mnew);
        sc[t][r] = p;
        psum += p;
      }
      for (int off = 8; off >= 1; off >>= 1)
        psum += __shfl_xor(psum, off, 16);
      lrow[r] = lrow[r] * corr + psum;
      for (int t = 0; t < 4; ++t) o[t][r] *= corr;
    }

    // relayout P (D-layout) -> LDS -> A-fragments
    h16* pp = Ps + w * 16 * 72;
    const int hl = lane >> 4, nc = lane & 15;
    for (int t = 0; t < 4; ++t)
      for (int r = 0; r < 8; ++r)
        pp[(r + hl * 8) * 72 + t * 16 + nc] = (h16)sc[t][r];
    __syncthreads();

    // O += P (16x64) x V_blk (64x64), B operand = V^T tile [d][n]
    v16h pa0 = load_frag_a(pp, 0, 0, 72, lane);
    v16h pa1 = load_frag_a(pp, 0, 32, 72, lane);
    for (int t = 0; t < 4; ++t) {
      v16h b0 = load_frag_b(Vc, t * 16, 0, 72, lane);
      v16h b1 = load_frag_b(Vc, t * 16, 32, 72, lane);
      o[t] = WMMA_F16(pa0, b0, o[t]);
      o[t] = WMMA_F16(pa1, b1, o[t]);
    }
    __syncthreads();  // all reads of buffer `cur` done before it is refilled
  }

  // normalize + write attention output in [token, C] layout, f16
  const int bi = bh / 12, hh = bh % 12;
  const int hl = lane >> 4, nc = lane & 15;
  for (int r = 0; r < 8; ++r) {
    float inv = 1.0f / lrow[r];
    int n = rb + w * 16 + r + hl * 8;
    size_t base = (size_t)(bi * 1024 + n) * 768 + hh * 64;
    for (int t = 0; t < 4; ++t)
      AO[base + t * 16 + nc] = (h16)(o[t][r] * inv);
  }
}

// ---- proj GEMM: [8192,768]f16 x [768,768]f16(B^T) + bias -> fp32 out -------
__global__ __launch_bounds__(128) void proj_gemm(const h16* __restrict__ X,
                                                 const h16* __restrict__ W,
                                                 const float* __restrict__ bias,
                                                 float* __restrict__ out) {
  __shared__ __align__(16) h16 As[64 * 40];
  __shared__ __align__(16) h16 Bs[64 * 40];
  const int tid = threadIdx.x, lane = tid & 31, wave = tid >> 5;
  const int wm = (wave >> 1) * 32, wn = (wave & 1) * 32;
  const int mb = blockIdx.x * 64;
  const int nb = blockIdx.y * 64;
  const int lr = tid >> 1, lk = (tid & 1) * 16;

  v8f acc[2][2] = {};
  for (int k0 = 0; k0 < 768; k0 += 32) {
    const v8h* ga = (const v8h*)(X + (size_t)(mb + lr) * 768 + k0 + lk);
    *(v8h*)(As + lr * 40 + lk)     = ga[0];
    *(v8h*)(As + lr * 40 + lk + 8) = ga[1];
    const v8h* gb = (const v8h*)(W + (size_t)(nb + lr) * 768 + k0 + lk);
    *(v8h*)(Bs + lr * 40 + lk)     = gb[0];
    *(v8h*)(Bs + lr * 40 + lk + 8) = gb[1];
    if (k0 + 32 < 768) {
      __builtin_prefetch(X + (size_t)(mb + lr) * 768 + k0 + 32 + lk, 0, 1);
      __builtin_prefetch(W + (size_t)(nb + lr) * 768 + k0 + 32 + lk, 0, 1);
    }
    __syncthreads();
    v16h a0 = load_frag_a(As, wm, 0, 40, lane);
    v16h a1 = load_frag_a(As, wm + 16, 0, 40, lane);
    v16h b0 = load_frag_b(Bs, wn, 0, 40, lane);
    v16h b1 = load_frag_b(Bs, wn + 16, 0, 40, lane);
    acc[0][0] = WMMA_F16(a0, b0, acc[0][0]);
    acc[0][1] = WMMA_F16(a0, b1, acc[0][1]);
    acc[1][0] = WMMA_F16(a1, b0, acc[1][0]);
    acc[1][1] = WMMA_F16(a1, b1, acc[1][1]);
    __syncthreads();
  }

  const int hl = lane >> 4, nc = lane & 15;
  for (int af = 0; af < 2; ++af)
    for (int tf = 0; tf < 2; ++tf)
      for (int r = 0; r < 8; ++r) {
        int m = mb + wm + af * 16 + r + hl * 8;
        int j = nb + wn + tf * 16 + nc;
        out[(size_t)m * 768 + j] = acc[af][tf][r] + bias[j];
      }
}

// ---------------------------------------------------------------------------
extern "C" void kernel_launch(void* const* d_in, const int* in_sizes, int n_in,
                              void* d_out, int out_size, void* d_ws, size_t ws_size,
                              hipStream_t stream) {
  (void)in_sizes; (void)n_in; (void)out_size; (void)ws_size;
  const float* xin   = (const float*)d_in[0];
  const float* yin   = (const float*)d_in[1];
  const float* wqkv  = (const float*)d_in[2];
  const float* wproj = (const float*)d_in[3];
  const float* bproj = (const float*)d_in[4];
  float* out = (float*)d_out;

  char* ws = (char*)d_ws;
  size_t off = 0;
  h16* Wq  = (h16*)(ws + off); off += (size_t)2304 * 768 * 2;
  h16* Wp  = (h16*)(ws + off); off += (size_t)768 * 768 * 2;
  h16* Xh  = (h16*)(ws + off); off += (size_t)8192 * 768 * 2;
  h16* Qb  = (h16*)(ws + off); off += (size_t)96 * 1024 * 64 * 2;
  h16* Kb  = (h16*)(ws + off); off += (size_t)96 * 1024 * 64 * 2;
  h16* Vtb = (h16*)(ws + off); off += (size_t)96 * 64 * 1024 * 2;
  h16* AO  = (h16*)(ws + off); off += (size_t)8192 * 768 * 2;  // ~67.6 MB total

  cvt4_f32_f16<<<(2304 * 768 / 4 + 255) / 256, 256, 0, stream>>>(wqkv, Wq, 2304 * 768 / 4);
  cvt4_f32_f16<<<(768 * 768 / 4 + 255) / 256, 256, 0, stream>>>(wproj, Wp, 768 * 768 / 4);

  for (int i = 0; i < 2; ++i) {
    const float* inp = (i == 0) ? xin : yin;
    cvt4_f32_f16<<<(8192 * 768 / 4 + 255) / 256, 256, 0, stream>>>(inp, Xh, 8192 * 768 / 4);
    qkv_gemm<<<dim3(128, 36), 128, 0, stream>>>(Xh, Wq, Qb, Kb, Vtb);
    attn_kernel<<<dim3(16, 96), 128, 0, stream>>>(Qb, Kb, Vtb, AO);
    proj_gemm<<<dim3(128, 12), 128, 0, stream>>>(AO, Wp, bproj,
                                                 out + (size_t)i * 8192 * 768);
  }
}
